// Model_25761213841886
// MI455X (gfx1250) — compile-verified, matrix operand order checked by reading, not run on hardware
//
#include <hip/hip_runtime.h>
#include <hip/hip_bf16.h>
#include <cstdint>
#include <cstddef>

typedef _Float16 h16;
typedef _Float16 v16h __attribute__((ext_vector_type(16)));
typedef _Float16 v8h  __attribute__((ext_vector_type(8)));
typedef float    v8f  __attribute__((ext_vector_type(8)));

// ---------------- kernels ----------------

// grid-stride zero fill (dwords)
__global__ void k_zero(uint32_t* __restrict__ p, size_t n) {
  size_t i = (size_t)blockIdx.x * blockDim.x + threadIdx.x;
  size_t st = (size_t)gridDim.x * blockDim.x;
  for (; i < n; i += st) p[i] = 0u;
}

// f32 -> f16 convert with row strides and dest column offset (padding stays 0)
__global__ void k_cvt(const float* __restrict__ src, int sld,
                      h16* __restrict__ dst, int dld, int doff,
                      int M, int K) {
  int i = blockIdx.x * blockDim.x + threadIdx.x;
  if (i >= M * K) return;
  int m = i / K, k = i - m * K;
  dst[(size_t)m * dld + doff + k] = (h16)src[(size_t)m * sld + k];
}

// seg_sum(x[dst] - x[src], dst): fp32 scatter-add
__global__ void k_seg_diff(const float* __restrict__ x, int ld,
                           const int* __restrict__ src, const int* __restrict__ dst,
                           int E, int D, float* __restrict__ acc) {
  int i = blockIdx.x * blockDim.x + threadIdx.x;
  if (i >= E * D) return;
  int e = i / D, d = i - e * D;
  int s = src[e], t = dst[e];
  float v = x[(size_t)t * ld + d] - x[(size_t)s * ld + d];
  atomicAdd(&acc[(size_t)t * D + d], v);
}

// seg_sum of +/- d_object_human: 8 obj pts x 25 hum pts x 3 = 600 dims
// value[(oi*25+hi)*3+c] = sign * (xh[hrow][hi*3+c] - xo[orow][oi*3+c])
__global__ void k_seg_ho(const float* __restrict__ xh, int ldh,
                         const float* __restrict__ xo, int ldo,
                         const int* __restrict__ eh, const int* __restrict__ eo,
                         int E, float sign, int acc_by_h,
                         float* __restrict__ acc) {
  int i = blockIdx.x * blockDim.x + threadIdx.x;
  if (i >= E * 600) return;
  int e = i / 600, r = i - e * 600;
  int oi = r / 75; int rr = r - oi * 75; int hi = rr / 3; int c = rr - hi * 3;
  int hrow = eh[e], orow = eo[e];
  float v = sign * (xh[(size_t)hrow * ldh + hi * 3 + c] -
                    xo[(size_t)orow * ldo + oi * 3 + c]);
  int row = acc_by_h ? hrow : orow;
  atomicAdd(&acc[(size_t)row * 600 + r], v);
}

// C[M,N] = A[M,Kp] @ B[N,Kp]^T via v_wmma_f32_16x16x32_f16.
// Each wave computes a 16x64 output strip: 1 M-tile x 4 N-tiles ->
//  - A fragment reused 4x per K-chunk (4 wmma : 10 b128 loads)
//  - 4 independent accumulator chains (hides WMMA pipeline latency)
// 4 waves/block. K multiple of 32 (zero-padded buffers). B rows padded to x16,
// so clamped duplicate tiles at the N edge read in-bounds (stores are guarded).
// A lane layout (16-bit A 16x32): lanes 0-15 row m, K={0..7,16..23};
// lanes 16-31 K={8..15,24..31}. B same pattern with lane -> n.
__global__ __launch_bounds__(128) void k_wmma_gemm(
    const h16* __restrict__ A, int lda,
    const h16* __restrict__ B, int ldb,
    float* __restrict__ C, int ldc,
    int M, int N, int K) {
  int lane = threadIdx.x;
  int numTN = (N + 15) >> 4;
  int tn0 = (blockIdx.x * 4 + threadIdx.y) * 4;  // first of 4 N-tiles (wave-uniform)
  if (tn0 >= numTN) return;                      // uniform: EXEC all-ones below
  int tm = blockIdx.y;
  int r  = lane & 15;
  int kb = (lane >> 4) << 3;                     // 0 or 8
  const h16* arow = A + (size_t)(tm * 16 + r) * lda + kb;
  int last = numTN - 1;
  int t1 = tn0 + 1 < numTN ? tn0 + 1 : last;
  int t2 = tn0 + 2 < numTN ? tn0 + 2 : last;
  int t3 = tn0 + 3 < numTN ? tn0 + 3 : last;
  const h16* b0p = B + (size_t)(tn0 * 16 + r) * ldb + kb;
  const h16* b1p = B + (size_t)(t1 * 16 + r) * ldb + kb;
  const h16* b2p = B + (size_t)(t2 * 16 + r) * ldb + kb;
  const h16* b3p = B + (size_t)(t3 * 16 + r) * ldb + kb;
  v8f acc0 = {0.f,0.f,0.f,0.f,0.f,0.f,0.f,0.f};
  v8f acc1 = acc0, acc2 = acc0, acc3 = acc0;
  for (int kk = 0; kk < K; kk += 32) {
    v8h a0 = *(const v8h*)(arow + kk);
    v8h a1 = *(const v8h*)(arow + kk + 16);
    v16h av;
#pragma unroll
    for (int j = 0; j < 8; ++j) { av[j] = a0[j]; av[j + 8] = a1[j]; }
    {
      v8h b0 = *(const v8h*)(b0p + kk), b1 = *(const v8h*)(b0p + kk + 16);
      v16h bv;
#pragma unroll
      for (int j = 0; j < 8; ++j) { bv[j] = b0[j]; bv[j + 8] = b1[j]; }
      acc0 = __builtin_amdgcn_wmma_f32_16x16x32_f16(false, av, false, bv,
                                                    (short)0, acc0, false, false);
    }
    {
      v8h b0 = *(const v8h*)(b1p + kk), b1 = *(const v8h*)(b1p + kk + 16);
      v16h bv;
#pragma unroll
      for (int j = 0; j < 8; ++j) { bv[j] = b0[j]; bv[j + 8] = b1[j]; }
      acc1 = __builtin_amdgcn_wmma_f32_16x16x32_f16(false, av, false, bv,
                                                    (short)0, acc1, false, false);
    }
    {
      v8h b0 = *(const v8h*)(b2p + kk), b1 = *(const v8h*)(b2p + kk + 16);
      v16h bv;
#pragma unroll
      for (int j = 0; j < 8; ++j) { bv[j] = b0[j]; bv[j + 8] = b1[j]; }
      acc2 = __builtin_amdgcn_wmma_f32_16x16x32_f16(false, av, false, bv,
                                                    (short)0, acc2, false, false);
    }
    {
      v8h b0 = *(const v8h*)(b3p + kk), b1 = *(const v8h*)(b3p + kk + 16);
      v16h bv;
#pragma unroll
      for (int j = 0; j < 8; ++j) { bv[j] = b0[j]; bv[j + 8] = b1[j]; }
      acc3 = __builtin_amdgcn_wmma_f32_16x16x32_f16(false, av, false, bv,
                                                    (short)0, acc3, false, false);
    }
  }
  // C/D layout: lane l -> n = l&15 ; VGPR v -> m = v + 8*(l>>4)
  int mbase = tm * 16 + ((lane >> 4) << 3);      // M always multiple of 16 here
  int nl = lane & 15;
  auto store_tile = [&](int tn, const v8f& a) {
    int n = tn * 16 + nl;
    if (n < N) {
#pragma unroll
      for (int v = 0; v < 8; ++v) C[(size_t)(mbase + v) * ldc + n] = a[v];
    }
  };
  store_tile(tn0, acc0);
  if (tn0 + 1 < numTN) store_tile(tn0 + 1, acc1);
  if (tn0 + 2 < numTN) store_tile(tn0 + 2, acc2);
  if (tn0 + 3 < numTN) store_tile(tn0 + 3, acc3);
}

// fused GRU pointwise: r,z,n gates + state update; writes f32 state, f16 state,
// and optionally an f16 copy into a concat buffer at column offset.
__global__ void k_gru_pw(const float* __restrict__ gi, const float* __restrict__ gh,
                         const float* __restrict__ bi, const float* __restrict__ bh,
                         float* __restrict__ h, h16* __restrict__ h16o,
                         h16* __restrict__ cat, int cat_ld, int cat_off,
                         int M, int H) {
  int i = blockIdx.x * blockDim.x + threadIdx.x;
  if (i >= M * H) return;
  int m = i / H, j = i - m * H;
  size_t g = (size_t)m * (3 * H);
  float ir = bi[j], iz = bi[H + j], in = bi[2 * H + j];
  if (gi) { ir += gi[g + j]; iz += gi[g + H + j]; in += gi[g + 2 * H + j]; }
  float hr = gh[g + j] + bh[j];
  float hz = gh[g + H + j] + bh[H + j];
  float hn = gh[g + 2 * H + j] + bh[2 * H + j];
  float rg = 1.0f / (1.0f + expf(-(ir + hr)));
  float zg = 1.0f / (1.0f + expf(-(iz + hz)));
  float ng = tanhf(in + rg * hn);
  size_t hi_ = (size_t)m * H + j;
  float hnew = (1.0f - zg) * ng + zg * h[hi_];
  h[hi_] = hnew;
  h16o[hi_] = (h16)hnew;
  if (cat) cat[(size_t)m * cat_ld + cat_off + j] = (h16)hnew;
}

// decode: vel = velraw + bias; y += vel; write y->d_out slot, y(f16)->concat col0,
// vel(f16)->vel16 buffer (input of th/to GRUs).
__global__ void k_y_update(const float* __restrict__ velraw, int ldv,
                           const float* __restrict__ bias,
                           float* __restrict__ y,
                           h16* __restrict__ cat0, int cat_ld,
                           h16* __restrict__ vel16, int v16ld,
                           float* __restrict__ out, int pred_len, int step,
                           int M, int D) {
  int i = blockIdx.x * blockDim.x + threadIdx.x;
  if (i >= M * D) return;
  int m = i / D, d = i - m * D;
  float v = velraw[(size_t)m * ldv + d] + bias[d];
  float yn = y[(size_t)m * D + d] + v;
  y[(size_t)m * D + d] = yn;
  cat0[(size_t)m * cat_ld + d] = (h16)yn;
  vel16[(size_t)m * v16ld + d] = (h16)v;
  out[((size_t)m * pred_len + step) * D + d] = yn;
}

// y = x[:, T-1, :]
__global__ void k_copy_last(const float* __restrict__ x, int T, int D,
                            float* __restrict__ y, int M) {
  int i = blockIdx.x * blockDim.x + threadIdx.x;
  if (i >= M * D) return;
  int m = i / D, d = i - m * D;
  y[i] = x[((size_t)m * T + (T - 1)) * D + d];
}

// ---------------- host ----------------

extern "C" void kernel_launch(void* const* d_in, const int* in_sizes, int n_in,
                              void* d_out, int out_size, void* d_ws, size_t ws_size,
                              hipStream_t stream) {
  // ---- problem constants (from setup_inputs) ----
  const int N_h = 512, N_o = 2048, T = 16;
  const int Dh = 75, Do = 24;
  const int E_hh = 4096, E_ho = 8192, E_oo = 32768;
  const int Hso = 256, Hsh = 128, Hoo = 128, Hth = 128, Hto = 128, Hnh = 256, Hno = 256;
  const int PRED = 12;
  // padded K dims (multiples of 32)
  const int KDh = 96, KDo = 32, K600 = 608, KnhI = 608 /*587*/, KnoI = 544 /*536*/;
  const int CAT_H = 608; // nh concat: xh(0:75) th(75:203) sho_h(203:459) shh(459:587)
  const int CAT_O = 544; // no concat: xo(0:24) to(24:152) sho_o(152:408) soo(408:536)

  // ---- inputs ----
  const float* x_h = (const float*)d_in[0];
  const float* x_o = (const float*)d_in[1];
  const int* hh_src = (const int*)d_in[2];
  const int* hh_dst = (const int*)d_in[3];
  const int* ho_src = (const int*)d_in[4];
  const int* ho_dst = (const int*)d_in[5];
  const int* oh_src = (const int*)d_in[6];
  const int* oh_dst = (const int*)d_in[7];
  const int* oo_src = (const int*)d_in[8];
  const int* oo_dst = (const int*)d_in[9];
  const float* sho_wi = (const float*)d_in[10]; const float* sho_wh = (const float*)d_in[11];
  const float* sho_bi = (const float*)d_in[12]; const float* sho_bh = (const float*)d_in[13];
  const float* shh_wi = (const float*)d_in[14]; const float* shh_wh = (const float*)d_in[15];
  const float* shh_bi = (const float*)d_in[16]; const float* shh_bh = (const float*)d_in[17];
  const float* soo_wi = (const float*)d_in[18]; const float* soo_wh = (const float*)d_in[19];
  const float* soo_bi = (const float*)d_in[20]; const float* soo_bh = (const float*)d_in[21];
  const float* th_wi  = (const float*)d_in[22]; const float* th_wh  = (const float*)d_in[23];
  const float* th_bi  = (const float*)d_in[24]; const float* th_bh  = (const float*)d_in[25];
  const float* to_wi  = (const float*)d_in[26]; const float* to_wh  = (const float*)d_in[27];
  const float* to_bi  = (const float*)d_in[28]; const float* to_bh  = (const float*)d_in[29];
  const float* nh_wi  = (const float*)d_in[30]; const float* nh_wh  = (const float*)d_in[31];
  const float* nh_bi  = (const float*)d_in[32]; const float* nh_bh  = (const float*)d_in[33];
  const float* no_wi  = (const float*)d_in[34]; const float* no_wh  = (const float*)d_in[35];
  const float* no_bi  = (const float*)d_in[36]; const float* no_bh  = (const float*)d_in[37];
  const float* outh_w = (const float*)d_in[38]; const float* outh_b = (const float*)d_in[39];
  const float* outo_w = (const float*)d_in[40]; const float* outo_b = (const float*)d_in[41];
  (void)in_sizes; (void)n_in; (void)out_size;

  float* out_h = (float*)d_out;                           // (N_h, PRED, Dh)
  float* out_o = (float*)d_out + (size_t)N_h * PRED * Dh; // (N_o, PRED, Do)

  // ---- workspace allocator ----
  char* base = (char*)d_ws;
  size_t off = 0;
  auto alloc = [&](size_t elems, size_t elsz) -> void* {
    off = (off + 255) & ~(size_t)255;
    void* p = base + off;
    off += elems * elsz;
    return p;
  };
  auto AH = [&](size_t n) { return (h16*)alloc(n, 2); };
  auto AF = [&](size_t n) { return (float*)alloc(n, 4); };

  // f16 weights (rows padded to x16, K padded to x32; pad regions stay zero)
  h16* w_sho_wi = AH((size_t)768 * K600);
  h16* w_sho_wh = AH((size_t)768 * 256);
  h16* w_shh_wi = AH((size_t)384 * KDh);
  h16* w_shh_wh = AH((size_t)384 * 128);
  h16* w_soo_wi = AH((size_t)384 * KDo);
  h16* w_soo_wh = AH((size_t)384 * 128);
  h16* w_th_wi  = AH((size_t)384 * KDh);
  h16* w_th_wh  = AH((size_t)384 * 128);
  h16* w_to_wi  = AH((size_t)384 * KDo);
  h16* w_to_wh  = AH((size_t)384 * 128);
  h16* w_nh_wi  = AH((size_t)768 * KnhI);
  h16* w_nh_wh  = AH((size_t)768 * 256);
  h16* w_no_wi  = AH((size_t)768 * KnoI);
  h16* w_no_wh  = AH((size_t)768 * 256);
  h16* w_outh   = AH((size_t)80 * 256);   // 75 rows valid
  h16* w_outo   = AH((size_t)32 * 256);   // 24 rows valid

  // concat sequences (encode) and single-step concat (decode)
  h16* nh_cat  = AH((size_t)T * N_h * CAT_H);
  h16* no_cat  = AH((size_t)T * N_o * CAT_O);
  h16* nh_catd = AH((size_t)N_h * CAT_H);
  h16* no_catd = AH((size_t)N_o * CAT_O);

  // segment-sum accumulators (f32) + f16 padded copies
  float* a_hh  = AF((size_t)N_h * Dh);
  float* a_oo  = AF((size_t)N_o * Do);
  float* a_hoo = AF((size_t)N_o * 600);
  float* a_hoh = AF((size_t)N_h * 600);
  h16* s_hh16  = AH((size_t)N_h * KDh);
  h16* s_oo16  = AH((size_t)N_o * KDo);
  h16* s_hoo16 = AH((size_t)N_o * K600);
  h16* s_hoh16 = AH((size_t)N_h * K600);

  // shared GEMM outputs
  float* gibuf = AF((size_t)N_o * 768);
  float* ghbuf = AF((size_t)N_o * 768);

  // GRU states f32 + f16
  float* h_shh  = AF((size_t)N_h * Hsh);  h16* g_shh  = AH((size_t)N_h * Hsh);
  float* h_shoo = AF((size_t)N_o * Hso);  h16* g_shoo = AH((size_t)N_o * Hso);
  float* h_shoh = AF((size_t)N_h * Hso);  h16* g_shoh = AH((size_t)N_h * Hso);
  float* h_soo  = AF((size_t)N_o * Hoo);  h16* g_soo  = AH((size_t)N_o * Hoo);
  float* h_th   = AF((size_t)N_h * Hth);  h16* g_th   = AH((size_t)N_h * Hth);
  float* h_to   = AF((size_t)N_o * Hto);  h16* g_to   = AH((size_t)N_o * Hto);
  float* h_nh   = AF((size_t)N_h * Hnh);  h16* g_nh   = AH((size_t)N_h * Hnh);
  float* h_no   = AF((size_t)N_o * Hno);  h16* g_no   = AH((size_t)N_o * Hno);

  // decode velocity + y
  float* velh = AF((size_t)N_h * 80);
  float* velo = AF((size_t)N_o * 32);
  h16* velh16 = AH((size_t)N_h * KDh);
  h16* velo16 = AH((size_t)N_o * KDo);
  float* y_h  = AF((size_t)N_h * Dh);
  float* y_o  = AF((size_t)N_o * Do);

  size_t used = (off + 3) & ~(size_t)3;
  if (used > ws_size) used = ws_size & ~(size_t)3;

  // ---- launch helpers ----
  auto zero = [&](void* p, size_t nf32) {
    unsigned b = (unsigned)((nf32 + 255) / 256); if (b > 4096) b = 4096;
    k_zero<<<b, 256, 0, stream>>>((uint32_t*)p, nf32);
  };
  auto cvt = [&](const float* s, int sld, h16* d, int dld, int doff, int M, int K) {
    int n = M * K;
    k_cvt<<<(n + 255) / 256, 256, 0, stream>>>(s, sld, d, dld, doff, M, K);
  };
  auto gemm = [&](const h16* A, int lda, const h16* B, int ldb, float* C, int ldc,
                  int M, int N, int K) {
    int tn = (N + 15) / 16;          // 16x16 tiles along N
    dim3 grid((tn + 15) / 16, M / 16, 1), blk(32, 4, 1);  // 4 waves x 4 tiles/wave
    k_wmma_gemm<<<grid, blk, 0, stream>>>(A, lda, B, ldb, C, ldc, M, N, K);
  };
  auto gru = [&](const float* gi, const float* gh, const float* bi, const float* bh,
                 float* h, h16* h16p, h16* cat, int cld, int coff, int M, int H) {
    int n = M * H;
    k_gru_pw<<<(n + 255) / 256, 256, 0, stream>>>(gi, gh, bi, bh, h, h16p, cat, cld, coff, M, H);
  };

  // ---- 0) zero the whole arena (states, pads, accumulators -> deterministic) ----
  k_zero<<<4096, 256, 0, stream>>>((uint32_t*)d_ws, used / 4);

  // ---- 1) convert weights to f16 (zero-padded) ----
  cvt(sho_wi, 600, w_sho_wi, K600, 0, 768, 600);
  cvt(sho_wh, 256, w_sho_wh, 256, 0, 768, 256);
  cvt(shh_wi, 75,  w_shh_wi, KDh, 0, 384, 75);
  cvt(shh_wh, 128, w_shh_wh, 128, 0, 384, 128);
  cvt(soo_wi, 24,  w_soo_wi, KDo, 0, 384, 24);
  cvt(soo_wh, 128, w_soo_wh, 128, 0, 384, 128);
  cvt(th_wi, 75,   w_th_wi,  KDh, 0, 384, 75);
  cvt(th_wh, 128,  w_th_wh,  128, 0, 384, 128);
  cvt(to_wi, 24,   w_to_wi,  KDo, 0, 384, 24);
  cvt(to_wh, 128,  w_to_wh,  128, 0, 384, 128);
  cvt(nh_wi, 587,  w_nh_wi,  KnhI, 0, 768, 587);
  cvt(nh_wh, 256,  w_nh_wh,  256, 0, 768, 256);
  cvt(no_wi, 536,  w_no_wi,  KnoI, 0, 768, 536);
  cvt(no_wh, 256,  w_no_wh,  256, 0, 768, 256);
  cvt(outh_w, 256, w_outh,   256, 0, 75, 256);
  cvt(outo_w, 256, w_outo,   256, 0, 24, 256);

  // ---- 2) x parts of the nh/no concat sequences ----
  for (int t = 0; t < T; ++t) {
    cvt(x_h + (size_t)t * Dh, T * Dh, nh_cat + (size_t)t * N_h * CAT_H, CAT_H, 0, N_h, Dh);
    cvt(x_o + (size_t)t * Do, T * Do, no_cat + (size_t)t * N_o * CAT_O, CAT_O, 0, N_o, Do);
  }

  // ---- 3) encode scans (each writes its f16 output into the concat buffers) ----
  // shh: seg_sum(x_h[dst]-x_h[src]) -> GRU(H=128) -> nh_cat col 459
  for (int t = 0; t < T; ++t) {
    zero(a_hh, (size_t)N_h * Dh);
    int n = E_hh * Dh;
    k_seg_diff<<<(n + 255) / 256, 256, 0, stream>>>(x_h + (size_t)t * Dh, T * Dh,
                                                    hh_src, hh_dst, E_hh, Dh, a_hh);
    cvt(a_hh, Dh, s_hh16, KDh, 0, N_h, Dh);
    gemm(s_hh16, KDh, w_shh_wi, KDh, gibuf, 384, N_h, 384, KDh);
    gemm(g_shh, Hsh, w_shh_wh, 128, ghbuf, 384, N_h, 384, 128);
    gru(gibuf, ghbuf, shh_bi, shh_bh, h_shh, g_shh,
        nh_cat + (size_t)t * N_h * CAT_H, CAT_H, 459, N_h, Hsh);
  }
  // sho_o: seg over objects of -(hum - obj) -> GRU(H=256) -> no_cat col 152
  for (int t = 0; t < T; ++t) {
    zero(a_hoo, (size_t)N_o * 600);
    int n = E_ho * 600;
    k_seg_ho<<<(n + 255) / 256, 256, 0, stream>>>(x_h + (size_t)t * Dh, T * Dh,
                                                  x_o + (size_t)t * Do, T * Do,
                                                  ho_src, ho_dst, E_ho, -1.0f, 0, a_hoo);
    cvt(a_hoo, 600, s_hoo16, K600, 0, N_o, 600);
    gemm(s_hoo16, K600, w_sho_wi, K600, gibuf, 768, N_o, 768, K600);
    gemm(g_shoo, Hso, w_sho_wh, 256, ghbuf, 768, N_o, 768, 256);
    gru(gibuf, ghbuf, sho_bi, sho_bh, h_shoo, g_shoo,
        no_cat + (size_t)t * N_o * CAT_O, CAT_O, 152, N_o, Hso);
  }
  // sho_h: seg over humans of (hum - obj) -> GRU(H=256, same weights) -> nh_cat col 203
  for (int t = 0; t < T; ++t) {
    zero(a_hoh, (size_t)N_h * 600);
    int n = E_ho * 600;
    k_seg_ho<<<(n + 255) / 256, 256, 0, stream>>>(x_h + (size_t)t * Dh, T * Dh,
                                                  x_o + (size_t)t * Do, T * Do,
                                                  oh_dst, oh_src, E_ho, 1.0f, 1, a_hoh);
    cvt(a_hoh, 600, s_hoh16, K600, 0, N_h, 600);
    gemm(s_hoh16, K600, w_sho_wi, K600, gibuf, 768, N_h, 768, K600);
    gemm(g_shoh, Hso, w_sho_wh, 256, ghbuf, 768, N_h, 768, 256);
    gru(gibuf, ghbuf, sho_bi, sho_bh, h_shoh, g_shoh,
        nh_cat + (size_t)t * N_h * CAT_H, CAT_H, 203, N_h, Hso);
  }
  // soo -> no_cat col 408
  for (int t = 0; t < T; ++t) {
    zero(a_oo, (size_t)N_o * Do);
    int n = E_oo * Do;
    k_seg_diff<<<(n + 255) / 256, 256, 0, stream>>>(x_o + (size_t)t * Do, T * Do,
                                                    oo_src, oo_dst, E_oo, Do, a_oo);
    cvt(a_oo, Do, s_oo16, KDo, 0, N_o, Do);
    gemm(s_oo16, KDo, w_soo_wi, KDo, gibuf, 384, N_o, 384, KDo);
    gemm(g_soo, Hoo, w_soo_wh, 128, ghbuf, 384, N_o, 384, 128);
    gru(gibuf, ghbuf, soo_bi, soo_bh, h_soo, g_soo,
        no_cat + (size_t)t * N_o * CAT_O, CAT_O, 408, N_o, Hoo);
  }
  // th/to: zero input => gi GEMM skipped (bias only) -> cat cols 75 / 24
  for (int t = 0; t < T; ++t) {
    gemm(g_th, Hth, w_th_wh, 128, ghbuf, 384, N_h, 384, 128);
    gru(nullptr, ghbuf, th_bi, th_bh, h_th, g_th,
        nh_cat + (size_t)t * N_h * CAT_H, CAT_H, 75, N_h, Hth);
  }
  for (int t = 0; t < T; ++t) {
    gemm(g_to, Hto, w_to_wh, 128, ghbuf, 384, N_o, 384, 128);
    gru(nullptr, ghbuf, to_bi, to_bh, h_to, g_to,
        no_cat + (size_t)t * N_o * CAT_O, CAT_O, 24, N_o, Hto);
  }
  // nh / no scans over the fully-built concat sequences
  for (int t = 0; t < T; ++t) {
    gemm(nh_cat + (size_t)t * N_h * CAT_H, CAT_H, w_nh_wi, KnhI, gibuf, 768, N_h, 768, KnhI);
    gemm(g_nh, Hnh, w_nh_wh, 256, ghbuf, 768, N_h, 768, 256);
    gru(gibuf, ghbuf, nh_bi, nh_bh, h_nh, g_nh, nullptr, 0, 0, N_h, Hnh);
  }
  for (int t = 0; t < T; ++t) {
    gemm(no_cat + (size_t)t * N_o * CAT_O, CAT_O, w_no_wi, KnoI, gibuf, 768, N_o, 768, KnoI);
    gemm(g_no, Hno, w_no_wh, 256, ghbuf, 768, N_o, 768, 256);
    gru(gibuf, ghbuf, no_bi, no_bh, h_no, g_no, nullptr, 0, 0, N_o, Hno);
  }

  // ---- 4) decode ----
  {
    int n = N_h * Dh;
    k_copy_last<<<(n + 255) / 256, 256, 0, stream>>>(x_h, T, Dh, y_h, N_h);
    n = N_o * Do;
    k_copy_last<<<(n + 255) / 256, 256, 0, stream>>>(x_o, T, Do, y_o, N_o);
  }

  for (int s = 0; s < PRED; ++s) {
    // vel = h_n @ out_w^T (+ bias fused into y_update)
    gemm(g_nh, Hnh, w_outh, 256, velh, 80, N_h, Dh, 256);
    gemm(g_no, Hno, w_outo, 256, velo, 32, N_o, Do, 256);
    {
      int n = N_h * Dh;
      k_y_update<<<(n + 255) / 256, 256, 0, stream>>>(velh, 80, outh_b, y_h,
                                                      nh_catd, CAT_H, velh16, KDh,
                                                      out_h, PRED, s, N_h, Dh);
      n = N_o * Do;
      k_y_update<<<(n + 255) / 256, 256, 0, stream>>>(velo, 32, outo_b, y_o,
                                                      no_catd, CAT_O, velo16, KDo,
                                                      out_o, PRED, s, N_o, Do);
    }
    // shh cell
    zero(a_hh, (size_t)N_h * Dh);
    {
      int n = E_hh * Dh;
      k_seg_diff<<<(n + 255) / 256, 256, 0, stream>>>(y_h, Dh, hh_src, hh_dst, E_hh, Dh, a_hh);
    }
    cvt(a_hh, Dh, s_hh16, KDh, 0, N_h, Dh);
    gemm(s_hh16, KDh, w_shh_wi, KDh, gibuf, 384, N_h, 384, KDh);
    gemm(g_shh, Hsh, w_shh_wh, 128, ghbuf, 384, N_h, 384, 128);
    gru(gibuf, ghbuf, shh_bi, shh_bh, h_shh, g_shh, nh_catd, CAT_H, 459, N_h, Hsh);
    // sho_o cell
    zero(a_hoo, (size_t)N_o * 600);
    {
      int n = E_ho * 600;
      k_seg_ho<<<(n + 255) / 256, 256, 0, stream>>>(y_h, Dh, y_o, Do,
                                                    ho_src, ho_dst, E_ho, -1.0f, 0, a_hoo);
    }
    cvt(a_hoo, 600, s_hoo16, K600, 0, N_o, 600);
    gemm(s_hoo16, K600, w_sho_wi, K600, gibuf, 768, N_o, 768, K600);
    gemm(g_shoo, Hso, w_sho_wh, 256, ghbuf, 768, N_o, 768, 256);
    gru(gibuf, ghbuf, sho_bi, sho_bh, h_shoo, g_shoo, no_catd, CAT_O, 152, N_o, Hso);
    // sho_h cell
    zero(a_hoh, (size_t)N_h * 600);
    {
      int n = E_ho * 600;
      k_seg_ho<<<(n + 255) / 256, 256, 0, stream>>>(y_h, Dh, y_o, Do,
                                                    oh_dst, oh_src, E_ho, 1.0f, 1, a_hoh);
    }
    cvt(a_hoh, 600, s_hoh16, K600, 0, N_h, 600);
    gemm(s_hoh16, K600, w_sho_wi, K600, gibuf, 768, N_h, 768, K600);
    gemm(g_shoh, Hso, w_sho_wh, 256, ghbuf, 768, N_h, 768, 256);
    gru(gibuf, ghbuf, sho_bi, sho_bh, h_shoh, g_shoh, nh_catd, CAT_H, 203, N_h, Hso);
    // soo cell
    zero(a_oo, (size_t)N_o * Do);
    {
      int n = E_oo * Do;
      k_seg_diff<<<(n + 255) / 256, 256, 0, stream>>>(y_o, Do, oo_src, oo_dst, E_oo, Do, a_oo);
    }
    cvt(a_oo, Do, s_oo16, KDo, 0, N_o, Do);
    gemm(s_oo16, KDo, w_soo_wi, KDo, gibuf, 384, N_o, 384, KDo);
    gemm(g_soo, Hoo, w_soo_wh, 128, ghbuf, 384, N_o, 384, 128);
    gru(gibuf, ghbuf, soo_bi, soo_bh, h_soo, g_soo, no_catd, CAT_O, 408, N_o, Hoo);
    // th cell: x = vel_h
    gemm(velh16, KDh, w_th_wi, KDh, gibuf, 384, N_h, 384, KDh);
    gemm(g_th, Hth, w_th_wh, 128, ghbuf, 384, N_h, 384, 128);
    gru(gibuf, ghbuf, th_bi, th_bh, h_th, g_th, nh_catd, CAT_H, 75, N_h, Hth);
    // to cell: x = vel_o
    gemm(velo16, KDo, w_to_wi, KDo, gibuf, 384, N_o, 384, KDo);
    gemm(g_to, Hto, w_to_wh, 128, ghbuf, 384, N_o, 384, 128);
    gru(gibuf, ghbuf, to_bi, to_bh, h_to, g_to, no_catd, CAT_O, 24, N_o, Hto);
    // nh cell on decode concat
    gemm(nh_catd, CAT_H, w_nh_wi, KnhI, gibuf, 768, N_h, 768, KnhI);
    gemm(g_nh, Hnh, w_nh_wh, 256, ghbuf, 768, N_h, 768, 256);
    gru(gibuf, ghbuf, nh_bi, nh_bh, h_nh, g_nh, nullptr, 0, 0, N_h, Hnh);
    // no cell on decode concat
    gemm(no_catd, CAT_O, w_no_wi, KnoI, gibuf, 768, N_o, 768, KnoI);
    gemm(g_no, Hno, w_no_wh, 256, ghbuf, 768, N_o, 768, 256);
    gru(gibuf, ghbuf, no_bi, no_bh, h_no, g_no, nullptr, 0, 0, N_o, Hno);
  }
}
